// CustomAttention_52372831208131
// MI455X (gfx1250) — matrix operand order repaired and checked by reading, not compiled, not run on hardware
//
#include <hip/hip_runtime.h>
#include <hip/hip_bf16.h>

typedef __attribute__((ext_vector_type(16))) __bf16 bf16x16;
typedef __attribute__((ext_vector_type(8)))  float  f32x8;
typedef unsigned long long u64;
typedef unsigned int u32;

// Problem sizes (fixed by the reference)
constexpr int B_  = 2;
constexpr int S_  = 2048;
constexpr int D_  = 4096;
constexpr int NH_ = 32;
constexpr int HD_ = 128;
constexpr int M_  = B_ * S_;   // 4096 rows for the projection GEMMs

// ---------------------------------------------------------------------------
// gfx1250 async copy: global -> LDS, 16 B per lane, tracked by ASYNCcnt.
// lds_off is the per-lane LDS byte offset (low 32 bits of the shared-aperture
// generic address); gptr is the per-lane global address.
// ---------------------------------------------------------------------------
__device__ __forceinline__ void async_copy_b128(u32 lds_off, const void* gptr) {
    asm volatile("global_load_async_to_lds_b128 %0, %1, off"
                 :: "v"(lds_off), "v"(gptr)
                 : "memory");
}
__device__ __forceinline__ void wait_asynccnt0() {
    asm volatile("s_wait_asynccnt 0" ::: "memory");
}
__device__ __forceinline__ u32 lds_off_of(const void* p) {
    return (u32)(uintptr_t)p;
}

// ---------------------------------------------------------------------------
// fp32 -> bf16 conversion (bandwidth op)
// ---------------------------------------------------------------------------
__global__ void cvt_f32_bf16(const float* __restrict__ in, __bf16* __restrict__ out, int n) {
    int i = blockIdx.x * blockDim.x + threadIdx.x;
    if (i < n) out[i] = (__bf16)in[i];
}

// ---------------------------------------------------------------------------
// RoPE: one thread per (even,odd) rotation pair.  Layout [B,S,NH,HD] means the
// flat element offset of pair idx is exactly 2*idx.
// ---------------------------------------------------------------------------
__global__ void rope_bf16(__bf16* __restrict__ qk,
                          const float* __restrict__ fcos,
                          const float* __restrict__ fsin) {
    u64 idx = (u64)blockIdx.x * blockDim.x + threadIdx.x;
    const u64 total = (u64)B_ * S_ * NH_ * (HD_ / 2);
    if (idx >= total) return;
    int i = (int)(idx % (HD_ / 2));
    u64 t = idx / (HD_ / 2);
    t /= NH_;                       // drop head index
    int s = (int)(t % S_);
    float c  = fcos[s * (HD_ / 2) + i];
    float sn = fsin[s * (HD_ / 2) + i];
    u64 base = idx * 2;
    float x0 = (float)qk[base];
    float x1 = (float)qk[base + 1];
    qk[base]     = (__bf16)(x0 * c - x1 * sn);
    qk[base + 1] = (__bf16)(x0 * sn + x1 * c);
}

// ---------------------------------------------------------------------------
// Tiled WMMA GEMM:  Out[M][N] = A[M][K] * W[N][K]^T
// Block tile 128x128, 8 waves, each wave -> 64x32 (4 M-subtiles x 2 N-subtiles).
// Double-buffered LDS stages filled by global_load_async_to_lds_b128; the
// end-of-iteration barrier (with its implicit dscnt wait) makes the one-barrier
// two-stage pipeline race-free.
// ---------------------------------------------------------------------------
template <bool WRITE_F32>
__global__ void __launch_bounds__(256, 1)
gemm_bf16_wmma(const __bf16* __restrict__ A,
               const __bf16* __restrict__ W,
               void* __restrict__ Out,
               int M, int N, int K) {
    constexpr int BM = 128, BN = 128, BK = 32;
    constexpr int LP = 40;                     // padded LDS row stride (elements)
    __shared__ __bf16 Asm[2][BM][LP];
    __shared__ __bf16 Bsm[2][BN][LP];

    const int tid  = threadIdx.x;
    const int wave = tid >> 5;
    const int lane = tid & 31;
    const int hl   = lane >> 4;                // half-wave select
    const int lrow = lane & 15;
    const int wy   = wave >> 2;                // 0..1 -> M offset 0/64
    const int wx   = wave & 3;                 // 0..3 -> N offset 0/32/64/96
    const int m0   = blockIdx.y * BM;
    const int n0   = blockIdx.x * BN;

    // per-thread staging slot: 32 B of one row of each tile
    const int sr = tid >> 1;                   // 0..127
    const int sc = (tid & 1) * 16;             // element col: 0 or 16

    f32x8 acc[4][2] = {};

    auto issue_stage = [&](int k0, int st) {
        const __bf16* ag = A + (u64)(m0 + sr) * K + k0 + sc;
        const __bf16* bg = W + (u64)(n0 + sr) * K + k0 + sc;
        u32 la = lds_off_of(&Asm[st][sr][sc]);
        u32 lb = lds_off_of(&Bsm[st][sr][sc]);
        async_copy_b128(la,      ag);
        async_copy_b128(la + 16, ag + 8);
        async_copy_b128(lb,      bg);
        async_copy_b128(lb + 16, bg + 8);
    };

    issue_stage(0, 0);
    wait_asynccnt0();
    __syncthreads();

    const int nk = K / BK;
    for (int kt = 0; kt < nk; ++kt) {
        const int cur = kt & 1;
        if (kt + 1 < nk) issue_stage((kt + 1) * BK, cur ^ 1);

        // ---- A fragments (ISA 16-bit 16x32 A layout) ----
        bf16x16 af[4];
#pragma unroll
        for (int mt = 0; mt < 4; ++mt) {
            int r = wy * 64 + mt * 16 + lrow;
#pragma unroll
            for (int dw = 0; dw < 8; ++dw) {
                int kk = (dw < 4 ? 0 : 16) + hl * 8 + (dw & 3) * 2;
                af[mt][2 * dw]     = Asm[cur][r][kk];
                af[mt][2 * dw + 1] = Asm[cur][r][kk + 1];
            }
        }
        // ---- B fragments: B[k][n] = W[n][k] (32x16 B layout) ----
        bf16x16 bfm[2];
#pragma unroll
        for (int nt = 0; nt < 2; ++nt) {
            int r = wx * 32 + nt * 16 + lrow;  // n index -> row of W tile
#pragma unroll
            for (int dw = 0; dw < 8; ++dw) {
                int kk = (dw < 4 ? 0 : 16) + hl * 8 + (dw & 3) * 2;
                bfm[nt][2 * dw]     = Bsm[cur][r][kk];
                bfm[nt][2 * dw + 1] = Bsm[cur][r][kk + 1];
            }
        }
#pragma unroll
        for (int mt = 0; mt < 4; ++mt)
#pragma unroll
            for (int nt = 0; nt < 2; ++nt)
                acc[mt][nt] = __builtin_amdgcn_wmma_f32_16x16x32_bf16(
                    false, af[mt], false, bfm[nt], (short)0, acc[mt][nt], false, false);

        wait_asynccnt0();   // next stage fully landed in LDS
        __syncthreads();    // everyone done reading cur (implicit dscnt wait)
    }

    // ---- store (C layout: lanes 0-15 N=lane M=v; lanes 16-31 N=lane-16 M=v+8)
#pragma unroll
    for (int mt = 0; mt < 4; ++mt)
#pragma unroll
        for (int nt = 0; nt < 2; ++nt)
#pragma unroll
            for (int v = 0; v < 8; ++v) {
                int m = m0 + wy * 64 + mt * 16 + hl * 8 + v;
                int n = n0 + wx * 32 + nt * 16 + lrow;
                float val = acc[mt][nt][v];
                if (WRITE_F32)
                    ((float*)Out)[(u64)m * N + n] = val;
                else
                    ((__bf16*)Out)[(u64)m * N + n] = (__bf16)val;
            }
}

// ---------------------------------------------------------------------------
// Flash attention (causal): block = 128 queries x one (b,h); 8 waves, each
// owns 16 query rows.  Q held as 4 resident A-fragments; 64-key tiles of K,V
// staged in LDS via async copies; online softmax; P re-laid out C->A via
// per-wave LDS buffer.
// ---------------------------------------------------------------------------
__global__ void __launch_bounds__(256, 1)
flash_attn_bf16(const __bf16* __restrict__ Q, const __bf16* __restrict__ K,
                const __bf16* __restrict__ V, __bf16* __restrict__ O) {
    constexpr int QB = 128, KB = 64;
    constexpr int KP = HD_ + 8;                // padded K/V row stride
    constexpr int PP = KB + 8;                 // padded P row stride
    __shared__ __bf16 Ksm[KB][KP];
    __shared__ __bf16 Vsm[KB][KP];
    __shared__ __bf16 Psm[8][16][PP];

    const int tid  = threadIdx.x;
    const int wave = tid >> 5;
    const int lane = tid & 31;
    const int hl   = lane >> 4;
    const int lrow = lane & 15;
    const int h    = blockIdx.y;
    const int b    = blockIdx.z;
    const int q0   = blockIdx.x * QB;
    const int qb   = q0 + wave * 16;           // this wave's first query row

    const u64 rowStride = (u64)NH_ * HD_;
    const u64 headOff   = (u64)h * HD_;
    const u64 batchOff  = (u64)b * S_ * rowStride;

    // resident Q fragments over d = 0..127 (4 x 16x32 A-frags)
    bf16x16 qf[4];
    {
        const __bf16* qrow = Q + batchOff + (u64)(qb + lrow) * rowStride + headOff;
#pragma unroll
        for (int f = 0; f < 4; ++f)
#pragma unroll
            for (int dw = 0; dw < 8; ++dw) {
                int d = 32 * f + (dw < 4 ? 0 : 16) + hl * 8 + (dw & 3) * 2;
                qf[f][2 * dw]     = qrow[d];
                qf[f][2 * dw + 1] = qrow[d + 1];
            }
    }

    const float scale = 0.08838834764831845f;  // 1/sqrt(128)
    float mrun[8], lrun[8];
    f32x8 oacc[8] = {};
#pragma unroll
    for (int v = 0; v < 8; ++v) { mrun[v] = -1e30f; lrun[v] = 0.f; }

    // per-thread staging slot: 64 B of one K row and one V row
    const int sr = tid >> 2;                   // 0..63
    const int sc = (tid & 3) * 32;             // element col: 0/32/64/96

    const int kt_end = (q0 + QB) / KB;         // causal: keys <= last query
    for (int kt = 0; kt < kt_end; ++kt) {
        const int k0 = kt * KB;
        // ---- async K/V tile staging: 64 rows x 256 B each ----
        {
            const __bf16* kg = K + batchOff + (u64)(k0 + sr) * rowStride + headOff + sc;
            const __bf16* vg = V + batchOff + (u64)(k0 + sr) * rowStride + headOff + sc;
            u32 lk = lds_off_of(&Ksm[sr][sc]);
            u32 lv = lds_off_of(&Vsm[sr][sc]);
#pragma unroll
            for (int i = 0; i < 4; ++i) {
                async_copy_b128(lk + i * 16, kg + i * 8);
                async_copy_b128(lv + i * 16, vg + i * 8);
            }
        }
        wait_asynccnt0();
        __syncthreads();

        // ---- scores: S = Q K^T  (4 key sub-tiles x 4 d-chunks) ----
        f32x8 sacc[4] = {};
#pragma unroll
        for (int f = 0; f < 4; ++f) {
            bf16x16 kf[4];
#pragma unroll
            for (int nt = 0; nt < 4; ++nt) {
                int kr = nt * 16 + lrow;       // key row within tile
#pragma unroll
                for (int dw = 0; dw < 8; ++dw) {
                    int d = 32 * f + (dw < 4 ? 0 : 16) + hl * 8 + (dw & 3) * 2;
                    kf[nt][2 * dw]     = Ksm[kr][d];
                    kf[nt][2 * dw + 1] = Ksm[kr][d + 1];
                }
            }
#pragma unroll
            for (int nt = 0; nt < 4; ++nt)
                sacc[nt] = __builtin_amdgcn_wmma_f32_16x16x32_bf16(
                    false, qf[f], false, kf[nt], (short)0, sacc[nt], false, false);
        }

        // ---- causal mask + online softmax (row = v in VGPR dim) ----
#pragma unroll
        for (int v = 0; v < 8; ++v) {
            int q = qb + hl * 8 + v;
            float mx = mrun[v];
#pragma unroll
            for (int nt = 0; nt < 4; ++nt) {
                int kc = k0 + nt * 16 + lrow;
                float s = (kc <= q) ? sacc[nt][v] * scale : -1e30f;
                sacc[nt][v] = s;
                mx = fmaxf(mx, s);
            }
#pragma unroll
            for (int off = 1; off < 16; off <<= 1)
                mx = fmaxf(mx, __shfl_xor(mx, off, 32));
            float rescale = __expf(mrun[v] - mx);
            lrun[v] *= rescale;
            float psum = 0.f;
#pragma unroll
            for (int nt = 0; nt < 4; ++nt) {
                float p = __expf(sacc[nt][v] - mx);
                sacc[nt][v] = p;
                psum += p;
            }
#pragma unroll
            for (int off = 1; off < 16; off <<= 1)
                psum += __shfl_xor(psum, off, 32);
            lrun[v] += psum;
            mrun[v] = mx;
#pragma unroll
            for (int d8 = 0; d8 < 8; ++d8) oacc[d8][v] *= rescale;
        }

        // ---- P: C-layout -> LDS -> A-layout ----
#pragma unroll
        for (int nt = 0; nt < 4; ++nt)
#pragma unroll
            for (int v = 0; v < 8; ++v)
                Psm[wave][hl * 8 + v][nt * 16 + lrow] = (__bf16)sacc[nt][v];

        bf16x16 pf[2];
#pragma unroll
        for (int f = 0; f < 2; ++f)
#pragma unroll
            for (int dw = 0; dw < 8; ++dw) {
                int kk = 32 * f + (dw < 4 ? 0 : 16) + hl * 8 + (dw & 3) * 2;
                pf[f][2 * dw]     = Psm[wave][lrow][kk];
                pf[f][2 * dw + 1] = Psm[wave][lrow][kk + 1];
            }

        // ---- O += P V  (8 d sub-tiles x 2 key-chunks) ----
#pragma unroll
        for (int f = 0; f < 2; ++f)
#pragma unroll
            for (int nt = 0; nt < 8; ++nt) {
                bf16x16 vf;
#pragma unroll
                for (int dw = 0; dw < 8; ++dw) {
                    int kk = 32 * f + (dw < 4 ? 0 : 16) + hl * 8 + (dw & 3) * 2;   // key index in tile
                    vf[2 * dw]     = Vsm[kk][nt * 16 + lrow];
                    vf[2 * dw + 1] = Vsm[kk + 1][nt * 16 + lrow];
                }
                oacc[nt] = __builtin_amdgcn_wmma_f32_16x16x32_bf16(
                    false, pf[f], false, vf, (short)0, oacc[nt], false, false);
            }
        __syncthreads();
    }

    // ---- finalize: divide by l, store bf16 [B,S,NH,HD] ----
#pragma unroll
    for (int nt = 0; nt < 8; ++nt)
#pragma unroll
        for (int v = 0; v < 8; ++v) {
            int q = qb + hl * 8 + v;
            int d = nt * 16 + lrow;
            float val = oacc[nt][v] / lrun[v];
            O[batchOff + (u64)q * rowStride + headOff + d] = (__bf16)val;
        }
}

// ---------------------------------------------------------------------------
// Host-side orchestration
// ---------------------------------------------------------------------------
extern "C" void kernel_launch(void* const* d_in, const int* in_sizes, int n_in,
                              void* d_out, int out_size, void* d_ws, size_t ws_size,
                              hipStream_t stream) {
    const float* x  = (const float*)d_in[0];
    const float* wq = (const float*)d_in[1];
    const float* wk = (const float*)d_in[2];
    const float* wv = (const float*)d_in[3];
    const float* wo = (const float*)d_in[4];
    const float* fc = (const float*)d_in[5];
    const float* fs = (const float*)d_in[6];
    // d_in[7] = mask (causal, computed analytically), d_in[8] = start_pos (0)

    char* ws = (char*)d_ws;
    const size_t MB = 1ull << 20;
    __bf16* xb  = (__bf16*)(ws + 0 * MB);
    __bf16* wqb = (__bf16*)(ws + 32 * MB);
    __bf16* wkb = (__bf16*)(ws + 64 * MB);
    __bf16* wvb = (__bf16*)(ws + 96 * MB);
    __bf16* wob = (__bf16*)(ws + 128 * MB);
    __bf16* Qb  = (__bf16*)(ws + 160 * MB);
    __bf16* Kb  = (__bf16*)(ws + 192 * MB);
    __bf16* Vb  = (__bf16*)(ws + 224 * MB);
    __bf16* Ab  = (__bf16*)(ws + 256 * MB);

    const int NE = M_ * D_;  // 16M elements per matrix
    cvt_f32_bf16<<<NE / 256, 256, 0, stream>>>(x,  xb,  NE);
    cvt_f32_bf16<<<NE / 256, 256, 0, stream>>>(wq, wqb, NE);
    cvt_f32_bf16<<<NE / 256, 256, 0, stream>>>(wk, wkb, NE);
    cvt_f32_bf16<<<NE / 256, 256, 0, stream>>>(wv, wvb, NE);
    cvt_f32_bf16<<<NE / 256, 256, 0, stream>>>(wo, wob, NE);

    dim3 gg(D_ / 128, M_ / 128);   // (32, 32)
    gemm_bf16_wmma<false><<<gg, 256, 0, stream>>>(xb, wqb, Qb, M_, D_, D_);
    gemm_bf16_wmma<false><<<gg, 256, 0, stream>>>(xb, wkb, Kb, M_, D_, D_);
    gemm_bf16_wmma<false><<<gg, 256, 0, stream>>>(xb, wvb, Vb, M_, D_, D_);

    const u64 pairs = (u64)B_ * S_ * NH_ * (HD_ / 2);  // 8M
    rope_bf16<<<(unsigned)(pairs / 256), 256, 0, stream>>>(Qb, fc, fs);
    rope_bf16<<<(unsigned)(pairs / 256), 256, 0, stream>>>(Kb, fc, fs);

    dim3 ga(S_ / 128, NH_, B_);    // (16, 32, 2)
    flash_attn_bf16<<<ga, 256, 0, stream>>>(Qb, Kb, Vb, Ab);

    gemm_bf16_wmma<true><<<gg, 256, 0, stream>>>(Ab, wob, d_out, M_, D_, D_);
}
